// LSTM_4_53979148976704
// MI455X (gfx1250) — matrix operand order repaired and checked by reading, compiled, not run on hardware
//
#include <hip/hip_runtime.h>
#include <hip/hip_bf16.h>

typedef __bf16 bf16_t;
typedef __attribute__((ext_vector_type(16))) __bf16 v16bf;
typedef __attribute__((ext_vector_type(8)))  __bf16 v8bf;
typedef __attribute__((ext_vector_type(8)))  float  v8f;

constexpr int kBatch = 4096;
constexpr int kSeq   = 8192;
constexpr int kHid   = 50;    // LSTM hidden
constexpr int kHPad  = 64;    // hidden padded to 4 x 16 WMMA tiles
constexpr int kRow   = 72;    // LDS row stride in bf16 (144B: 16B aligned, bank-skewed)
constexpr int kFc1   = 150;
constexpr int kOut   = 2;

__device__ __forceinline__ float fexp(float x) {
  return __builtin_amdgcn_exp2f(x * 1.44269504088896340736f);   // v_exp_f32
}
__device__ __forceinline__ float fsig(float x) {
  return __builtin_amdgcn_rcpf(1.0f + fexp(-x));                // v_rcp_f32
}
__device__ __forceinline__ float ftanh(float x) {
  return 2.0f * __builtin_amdgcn_rcpf(1.0f + fexp(-2.0f * x)) - 1.0f;
}

// One block = 16 batch rows, 4 waves. Wave w owns hidden slice [16w,16w+16)
// across all four gates (i,f,g,o), so the cell update is lane-local.
// Weights (W_hh, padded 50->64 with zeros) are resident in VGPRs as WMMA
// B-fragments; h is exchanged between waves via a double-buffered LDS tile.
__global__ __launch_bounds__(128, 1)
void lstm_scan_kernel(const float* __restrict__ xs,    // seq flat: [t*4096 + b]
                      const float* __restrict__ h0,    // [B*50]
                      const float* __restrict__ c0,    // [B*50]
                      const float* __restrict__ W_ih,  // [200]
                      const float* __restrict__ W_hh,  // [200*50]
                      const float* __restrict__ b_ih,  // [200]
                      const float* __restrict__ b_hh,  // [200]
                      float* __restrict__ h_last)      // [B*kHPad]
{
  __shared__ bf16_t hbuf[2][16 * kRow];

  const int lane = threadIdx.x & 31;
  const int wave = threadIdx.x >> 5;      // hidden-slice 0..3
  const int half = lane >> 4;             // lane half (WMMA layout)
  const int nloc = lane & 15;             // N index within tile
  const int jh   = wave * 16 + nloc;      // padded hidden index 0..63
  const bool jv  = (jh < kHid);
  const int b0   = blockIdx.x * 16;       // batch tile base

  // Per-lane constants: combined bias and input weight for each gate
  // (INPUT_SIZE==1 -> the whole input projection is x[b]*W_ih[g] + bias).
  float biasq[4], wihq[4];
#pragma unroll
  for (int q = 0; q < 4; ++q) {
    biasq[q] = jv ? (b_ih[q * kHid + jh] + b_hh[q * kHid + jh]) : 0.0f;
    wihq[q]  = jv ? W_ih[q * kHid + jh] : 0.0f;
  }

  // Resident B fragments: B(k,n) = W_hh[n_global, k], n_global = q*50 + jh.
  // 16-bit B layout: lane L, element e -> N = L&15, K = (L>>4)*8 + e for e<8, 16 + (L>>4)*8 + (e-8) for e>=8 (+32 per chunk).
  v16bf bfrag[4][2];
#pragma unroll
  for (int q = 0; q < 4; ++q)
#pragma unroll
    for (int cc = 0; cc < 2; ++cc) {
      v16bf f;
#pragma unroll
      for (int e = 0; e < 16; ++e) {
        const int k = cc * 32 + (e < 8 ? half * 8 + e : 16 + half * 8 + (e - 8));
        const float w = (jv && k < kHid) ? W_hh[(q * kHid + jh) * kHid + k] : 0.0f;
        f[e] = (bf16_t)w;
      }
      bfrag[q][cc] = f;
    }

  // State in D-tile layout: lane (half,nloc), reg i -> (batch b0+half*8+i, hidden jh)
  float cst[8], hnew[8];
#pragma unroll
  for (int i = 0; i < 8; ++i) {
    const int b = b0 + half * 8 + i;
    cst[i]  = jv ? c0[b * kHid + jh] : 0.0f;
    const float h = jv ? h0[b * kHid + jh] : 0.0f;
    hbuf[0][(half * 8 + i) * kRow + jh] = (bf16_t)h;
    hnew[i] = h;
  }
  __syncthreads();

  int pb = 0;
  for (int t = 0; t < kSeq; ++t) {
    // A fragment: A(m,k) = h[b0+m, k]. 16-bit A layout: lane L -> M = L&15,
    // VGPRs 0-3 hold K = half*8..+8, VGPRs 4-7 hold K = 16+half*8..+8 (per chunk).
    const bf16_t* hb = hbuf[pb];
    const int ar = nloc * kRow + half * 8;
    union { v16bf w; v8bf p[2]; } a0u, a1u;
    a0u.p[0] = *(const v8bf*)&hb[ar];
    a0u.p[1] = *(const v8bf*)&hb[ar + 16];
    a1u.p[0] = *(const v8bf*)&hb[ar + 32];
    a1u.p[1] = *(const v8bf*)&hb[ar + 48];

    // x_t for this block's 16 batch rows (batch-contiguous, coalesced).
    const float* xp = xs + (size_t)t * kBatch + b0 + half * 8;
    const float4 xa = *(const float4*)xp;
    const float4 xb = *(const float4*)(xp + 4);
    const float xr[8] = {xa.x, xa.y, xa.z, xa.w, xb.x, xb.y, xb.z, xb.w};

    if (t + 8 < kSeq)
      __builtin_prefetch(xs + (size_t)(t + 8) * kBatch + b0, 0, 1);

    // gates[q] = h @ W_hh_q^T + x*W_ih_q + bias  (f32 accumulate, K=64 in 2 WMMAs)
    v8f g[4];
#pragma unroll
    for (int q = 0; q < 4; ++q) {
      v8f cin;
#pragma unroll
      for (int i = 0; i < 8; ++i) cin[i] = fmaf(xr[i], wihq[q], biasq[q]);
      v8f acc = __builtin_amdgcn_wmma_f32_16x16x32_bf16(
          false, a0u.w, false, bfrag[q][0], (short)0, cin, false, false);
      g[q] = __builtin_amdgcn_wmma_f32_16x16x32_bf16(
          false, a1u.w, false, bfrag[q][1], (short)0, acc, false, false);
    }

    // Elementwise LSTM cell (f32), write new h (bf16) into the other buffer.
    bf16_t* ho = hbuf[pb ^ 1];
#pragma unroll
    for (int i = 0; i < 8; ++i) {
      const float iv = fsig(g[0][i]);
      const float fv = fsig(g[1][i]);
      const float gv = ftanh(g[2][i]);
      const float ov = fsig(g[3][i]);
      const float cv = fmaf(fv, cst[i], iv * gv);
      cst[i] = cv;
      const float hv = ov * ftanh(cv);
      hnew[i] = hv;
      ho[(half * 8 + i) * kRow + jh] = (bf16_t)hv;
    }
    __syncthreads();   // single barrier per step (double-buffered h)
    pb ^= 1;
  }

#pragma unroll
  for (int i = 0; i < 8; ++i)
    h_last[(size_t)(b0 + half * 8 + i) * kHPad + jh] = hnew[i];
}

// Tiny FC head: relu(relu(h @ fc1^T + b1) @ fc2^T + b2). Weight indices are
// wave-uniform -> scalar loads; one thread per batch row.
__global__ __launch_bounds__(128)
void lstm_head_kernel(const float* __restrict__ hseq,   // [B*kHPad]
                      const float* __restrict__ fc1_w,  // [150*50]
                      const float* __restrict__ fc1_b,  // [150]
                      const float* __restrict__ fc2_w,  // [2*150]
                      const float* __restrict__ fc2_b,  // [2]
                      float* __restrict__ out)          // [B*2]
{
  const int b = blockIdx.x * blockDim.x + threadIdx.x;
  if (b >= kBatch) return;
  float h[kHid];
#pragma unroll
  for (int k = 0; k < kHid; ++k) h[k] = hseq[(size_t)b * kHPad + k];
  float o0 = 0.0f, o1 = 0.0f;
  for (int j = 0; j < kFc1; ++j) {
    float a = fc1_b[j];
#pragma unroll
    for (int k = 0; k < kHid; ++k) a = fmaf(h[k], fc1_w[j * kHid + k], a);
    a = fmaxf(a, 0.0f);
    o0 = fmaf(a, fc2_w[j], o0);
    o1 = fmaf(a, fc2_w[kFc1 + j], o1);
  }
  out[(size_t)b * kOut + 0] = fmaxf(o0 + fc2_b[0], 0.0f);
  out[(size_t)b * kOut + 1] = fmaxf(o1 + fc2_b[1], 0.0f);
}

extern "C" void kernel_launch(void* const* d_in, const int* in_sizes, int n_in,
                              void* d_out, int out_size, void* d_ws, size_t ws_size,
                              hipStream_t stream) {
  (void)in_sizes; (void)n_in; (void)out_size; (void)ws_size;
  const float* seq   = (const float*)d_in[0];
  const float* h0    = (const float*)d_in[1];
  const float* c0    = (const float*)d_in[2];
  const float* W_ih  = (const float*)d_in[3];
  const float* W_hh  = (const float*)d_in[4];
  const float* b_ih  = (const float*)d_in[5];
  const float* b_hh  = (const float*)d_in[6];
  const float* fc1_w = (const float*)d_in[7];
  const float* fc1_b = (const float*)d_in[8];
  const float* fc2_w = (const float*)d_in[9];
  const float* fc2_b = (const float*)d_in[10];

  float* hlast = (float*)d_ws;               // 4096 * 64 floats = 1 MiB scratch
  float* out   = (float*)d_out;

  lstm_scan_kernel<<<kBatch / 16, 128, 0, stream>>>(
      seq, h0, c0, W_ih, W_hh, b_ih, b_hh, hlast);
  lstm_head_kernel<<<(kBatch + 127) / 128, 128, 0, stream>>>(
      hlast, fc1_w, fc1_b, fc2_w, fc2_b, out);
}
